// ModulatedConv2d_63367947485948
// MI455X (gfx1250) — matrix operand order repaired and checked by reading, not compiled, not run on hardware
//
#include <hip/hip_runtime.h>
#include <hip/hip_bf16.h>

typedef _Float16 half_t;
typedef __attribute__((ext_vector_type(8)))  _Float16 v8h;
typedef __attribute__((ext_vector_type(16))) _Float16 v16h;
typedef __attribute__((ext_vector_type(8)))  float    v8f;

#define N_B    8
#define C_IN   64
#define C_OUT  64
#define HW     262144   /* 512*512 */
#define STYLE  512

// ---------------------------------------------------------------------------
// Kernel 1: style modulation + demodulation, folded once into f16 weights.
// grid = (8), block = (64).  Output: wt16[n][co][ci] row-major, 64 KB in d_ws.
// ---------------------------------------------------------------------------
__global__ __launch_bounds__(64)
void modw_kernel(const float* __restrict__ style,
                 const float* __restrict__ weight,
                 const float* __restrict__ w_mod,
                 const float* __restrict__ b_mod,
                 half_t* __restrict__ wt16) {
  __shared__ float s_s[C_IN];
  const int n = blockIdx.x;
  const int t = threadIdx.x;

  // s[ci] = style[n] . w_mod[ci] + b_mod[ci]
  {
    const float* st = style + n * STYLE;
    const float* wm = w_mod + t * STYLE;
    float acc = 0.f;
    #pragma unroll 8
    for (int k = 0; k < STYLE; ++k) acc = fmaf(st[k], wm[k], acc);
    s_s[t] = acc + b_mod[t];
  }
  __syncthreads();

  // Row co = t: modulate, demodulate, emit f16.
  const float* wrow = weight + t * C_IN;
  float sumsq = 0.f;
  #pragma unroll
  for (int ci = 0; ci < C_IN; ++ci) {
    float v = wrow[ci] * s_s[ci];
    sumsq = fmaf(v, v, sumsq);
  }
  const float d = rsqrtf(sumsq + 1e-8f);
  half_t* dst = wt16 + (n * C_OUT + t) * C_IN;
  #pragma unroll
  for (int ci = 0; ci < C_IN; ++ci) {
    dst[ci] = (half_t)(wrow[ci] * s_s[ci] * d);
  }
}

// ---------------------------------------------------------------------------
// A-fragment loader: 16-bit A 16x32 layout (ISA 7.12.2).
// Lane (m, hi): halfs 0-7 = row[hi*8 + 0..7], halfs 8-15 = row[16 + hi*8 + 0..7]
// -> exactly two 16-byte (v8h / b128) loads per fragment.
// ---------------------------------------------------------------------------
__device__ __forceinline__ v16h load_a_frag(const half_t* __restrict__ row, int hi) {
  const v8h lo = *(const v8h*)(row + hi * 8);
  const v8h up = *(const v8h*)(row + 16 + hi * 8);
  v16h a;
  #pragma unroll
  for (int i = 0; i < 8; ++i) { a[i] = lo[i]; a[i + 8] = up[i]; }
  return a;
}

// ---------------------------------------------------------------------------
// Kernel 2: per-sample 64x64 @ 64x262144 GEMM via v_wmma_f32_16x16x32_f16.
// grid = (HW/128, N), block = 256 (8 waves).
// Each wave owns ONE 16-pixel strip and ALL 64 output channels:
//   - B fragments (x) loaded once (non-temporal: x is single-use globally),
//   - reused across 4 co-tiles -> 8 WMMAs into 4 f32 accumulators,
//   - out written with non-temporal stores (write-once stream).
// ---------------------------------------------------------------------------
__global__ __launch_bounds__(256)
void conv1x1_wmma_kernel(const float* __restrict__ x,
                         const half_t* __restrict__ wt16,
                         float* __restrict__ out) {
  const int n    = blockIdx.y;
  const int lane = threadIdx.x & 31;
  const int wave = threadIdx.x >> 5;                    // 0..7
  const int p_base = blockIdx.x * 128 + wave * 16;      // 16-pixel strip

  const int m  = lane & 15;   // A row / B col / C col
  const int hi = lane >> 4;   // lane-half selector

  // ---- B fragments (16-bit B 32x16 layout): halfs 0-7 K=hi*8+0..7, halfs 8-15 K=16+hi*8+0..7
  // x is read exactly once across the whole grid -> TH=NT load.
  const float* xcol = x + ((size_t)n * C_IN) * (size_t)HW + (size_t)(p_base + m);
  v16h b0, b1;
  #pragma unroll
  for (int h = 0; h < 16; ++h) {
    const int kl = (h >> 3) * 16 + hi * 8 + (h & 7);
    b0[h] = (half_t)__builtin_nontemporal_load(xcol + (size_t)kl * (size_t)HW);
    b1[h] = (half_t)__builtin_nontemporal_load(xcol + (size_t)(kl + 32) * (size_t)HW);
  }

  // ---- A fragments: 4 co-tiles x 2 K-chunks, two b128 loads each ----------
  const half_t* wbase = wt16 + ((size_t)n * C_OUT) * C_IN + (size_t)m * C_IN;
  v8f c[4] = {v8f{}, v8f{}, v8f{}, v8f{}};
  #pragma unroll
  for (int ct = 0; ct < 4; ++ct) {
    const half_t* wrow = wbase + (size_t)(ct * 16) * C_IN;  // co = ct*16 + m
    const v16h a0 = load_a_frag(wrow, hi);
    const v16h a1 = load_a_frag(wrow + 32, hi);
    c[ct] = __builtin_amdgcn_wmma_f32_16x16x32_f16(false, a0, false, b0,
                                                   (short)0, c[ct], false, false);
    c[ct] = __builtin_amdgcn_wmma_f32_16x16x32_f16(false, a1, false, b1,
                                                   (short)0, c[ct], false, false);
  }

  // ---- Store (f32 C/D 16x16 layout: VGPR r -> rows r and r+8) -------------
  // out is write-once -> TH=NT stores; lanes 0-15 contiguous (64B/half-wave).
  #pragma unroll
  for (int ct = 0; ct < 4; ++ct) {
    float* orow = out + ((size_t)(n * C_OUT + ct * 16 + hi * 8)) * (size_t)HW
                      + (size_t)(p_base + m);
    #pragma unroll
    for (int r = 0; r < 8; ++r) {
      __builtin_nontemporal_store(c[ct][r], orow + (size_t)r * (size_t)HW);
    }
  }
}

// ---------------------------------------------------------------------------
extern "C" void kernel_launch(void* const* d_in, const int* in_sizes, int n_in,
                              void* d_out, int out_size, void* d_ws, size_t ws_size,
                              hipStream_t stream) {
  const float* x      = (const float*)d_in[0];
  const float* style  = (const float*)d_in[1];
  const float* weight = (const float*)d_in[2];
  const float* w_mod  = (const float*)d_in[3];
  const float* b_mod  = (const float*)d_in[4];
  float* out = (float*)d_out;
  half_t* wt16 = (half_t*)d_ws;   // 8*64*64 halfs = 64 KB

  modw_kernel<<<dim3(N_B), dim3(64), 0, stream>>>(style, weight, w_mod, b_mod, wt16);
  conv1x1_wmma_kernel<<<dim3(HW / 128, N_B), dim3(256), 0, stream>>>(x, wt16, out);
}
